// GPT_39247411150861
// MI455X (gfx1250) — matrix operand order repaired and checked by reading, not compiled, not run on hardware
//
#include <hip/hip_runtime.h>
#include <hip/hip_bf16.h>

// ---------------------------------------------------------------------------
// Types for CDNA5 WMMA (wave32): v_wmma_f32_16x16x32_f16
// ---------------------------------------------------------------------------
typedef _Float16 half16 __attribute__((ext_vector_type(16)));
typedef _Float16 half8  __attribute__((ext_vector_type(8)));
typedef float    v8f    __attribute__((ext_vector_type(8)));

union H16 { half16 v; half8 h[2]; _Float16 e[16]; };

static __device__ __forceinline__ v8f wmma_f16(half16 a, half16 b, v8f c) {
    // (neg_a, A, neg_b, B, c_mod, C, reuse_a, reuse_b)
    return __builtin_amdgcn_wmma_f32_16x16x32_f16(false, a, false, b, (short)0, c, false, false);
}

// Async global->LDS copy (gfx1250, ASYNCcnt-tracked). Per ISA §15.18.3:
// each lane copies 16 bytes from global to the LDS address in its VGPR, tracked with ASYNCcnt.
static __device__ __forceinline__ void async_copy_b128(unsigned lds_off, const _Float16* gptr) {
    unsigned long long ga = (unsigned long long)gptr;
    asm volatile("global_load_async_to_lds_b128 %0, %1, off"
                 :: "v"(lds_off), "v"(ga) : "memory");
}
static __device__ __forceinline__ void wait_async_le2() {
    asm volatile("s_wait_asynccnt 0x2" ::: "memory");
}
static __device__ __forceinline__ void wait_async_0() {
    asm volatile("s_wait_asynccnt 0x0" ::: "memory");
}

// Problem sizes (compile-time)
#define BB   16
#define TT   512
#define CC   1024
#define HH   16
#define DH   64
#define LL   8
#define FF   4096
#define MM   (BB * TT)   // 8192 rows in the big GEMMs
#define NOUT 256         // 2*LAT

enum { MODE_F16 = 0, MODE_VT = 1, MODE_RESID = 2, MODE_GELU = 3 };

// ---------------------------------------------------------------------------
// Weight packing: fp32 [K,N] row-major -> f16 WMMA B-fragments.
// Fragment (nt, kt) holds a 32(K) x 16(N) tile. This code's k order: lane l -> column
// n = nt*16 + (l&15); lane halves: l<16 -> K=kt*32+0..15, l>=16 -> +16..31,
// ascending within the lane's 16 contiguous f16 values.
// ---------------------------------------------------------------------------
__global__ __launch_bounds__(256) void pack_kernel(const float* __restrict__ W,
                                                   _Float16* __restrict__ P,
                                                   int K, int N) {
    int t    = blockIdx.x * 256 + threadIdx.x;
    int lane = t & 31;
    int frag = t >> 5;
    int ktc  = K >> 5;                 // k-tiles of 32
    int kt   = frag % ktc;
    int nt   = frag / ktc;
    int n    = nt * 16 + (lane & 15);
    int kb   = kt * 32 + (lane >> 4) * 16;
    const float* src = W + (size_t)kb * N + n;
    H16 o;
#pragma unroll
    for (int j = 0; j < 16; ++j) o.e[j] = (_Float16)src[(size_t)j * N];
    *(half16*)(P + (size_t)t * 16) = o.v;
}

// ---------------------------------------------------------------------------
// Tiled WMMA GEMM: out = epilogue(A_f16[M,K] @ Bpacked[K,N] + bias[N])
// Block: 256 threads (8 waves), tile 128x128, K-step 32.
// X tile staged via double-buffered async global->LDS copies (ASYNCcnt),
// issued one K-step ahead; B fragments streamed from the packed buffer.
// ---------------------------------------------------------------------------
__global__ __launch_bounds__(256) void gemm_kernel(const _Float16* __restrict__ A,
                                                   const _Float16* __restrict__ Bp,
                                                   const float* __restrict__ bias,
                                                   int K, int N, int mode,
                                                   float* __restrict__ out32,
                                                   _Float16* __restrict__ out16) {
    const int tid = threadIdx.x, w = tid >> 5, lane = tid & 31;
    const int g = lane >> 4, ln16 = lane & 15;
    const int wm = w >> 1, wn = w & 1;
    const int m0 = blockIdx.x * 128, n0 = blockIdx.y * 128;

    __shared__ _Float16 Xs[2][128 * 40];   // 32 k-values per row, stride 40 (pad)

    v8f acc[2][4];
#pragma unroll
    for (int i = 0; i < 2; ++i)
#pragma unroll
        for (int j = 0; j < 4; ++j) acc[i][j] = (v8f){0,0,0,0,0,0,0,0};

    const int ktc = K >> 5;
    const int lr = tid >> 1, lh = tid & 1;                 // tile loader: row, k-half
    const _Float16* Arow = A + (size_t)(m0 + lr) * K + lh * 16;
    unsigned lds_dst[2];
    lds_dst[0] = (unsigned)(unsigned long long)(&Xs[0][lr * 40 + lh * 16]);
    lds_dst[1] = (unsigned)(unsigned long long)(&Xs[1][lr * 40 + lh * 16]);

    const half16* bp16   = (const half16*)Bp;
    const size_t  fbase0 = (size_t)(blockIdx.y * 8 + wn * 4) * ktc;

    // prologue: async-stage tile 0 into buffer 0
    async_copy_b128(lds_dst[0],      Arow);
    async_copy_b128(lds_dst[0] + 16, Arow + 8);

    for (int kt = 0; kt < ktc; ++kt) {
        const int cur = kt & 1;
        if (kt + 1 < ktc) {
            const _Float16* nsrc = Arow + (size_t)(kt + 1) * 32;
            async_copy_b128(lds_dst[cur ^ 1],      nsrc);
            async_copy_b128(lds_dst[cur ^ 1] + 16, nsrc + 8);
            wait_async_le2();          // tile kt resident, tile kt+1 in flight
        } else {
            wait_async_0();            // last tile: drain
        }
        __syncthreads();

        // A fragments from LDS (padded: conflict-free b128 reads)
        H16 a0, a1;
        const _Float16* xr0 = &Xs[cur][(wm * 32 + ln16) * 40];
        const _Float16* xr1 = &Xs[cur][(wm * 32 + 16 + ln16) * 40];
        a0.h[0] = *(const half8*)(xr0 + g * 16);
        a0.h[1] = *(const half8*)(xr0 + g * 16 + 8);
        a1.h[0] = *(const half8*)(xr1 + g * 16);
        a1.h[1] = *(const half8*)(xr1 + g * 16 + 8);

        // B fragments straight from packed weights (32B/lane, coalesced)
        half16 bfr[4];
#pragma unroll
        for (int j = 0; j < 4; ++j) {
            size_t fi = (fbase0 + (size_t)j * ktc + kt) * 32 + lane;
            bfr[j] = bp16[fi];
            __builtin_prefetch(bp16 + fi + 32, 0, 1);   // next k-step
        }
#pragma unroll
        for (int j = 0; j < 4; ++j) {
            acc[0][j] = wmma_f16(a0.v, bfr[j], acc[0][j]);
            acc[1][j] = wmma_f16(a1.v, bfr[j], acc[1][j]);
        }
        __syncthreads();   // protect buffer reuse (next async overwrites cur^1... after one more step)
    }

    // epilogue: D tile element r,lane -> (M=(lane/16)*8+r, N=lane&15)
#pragma unroll
    for (int i = 0; i < 2; ++i) {
#pragma unroll
        for (int j = 0; j < 4; ++j) {
            int mb = m0 + wm * 32 + i * 16 + g * 8;
            int n  = n0 + wn * 64 + j * 16 + ln16;
            float bv = bias[n];
#pragma unroll
            for (int r = 0; r < 8; ++r) {
                int m = mb + r;
                float val = acc[i][j][r] + bv;
                if (mode == MODE_F16) {
                    out16[(size_t)m * N + n] = (_Float16)val;
                } else if (mode == MODE_VT) {      // V -> [B, C(=h*D+d), T]
                    int bi = m >> 9, t = m & 511;
                    out16[((size_t)bi * N + n) * TT + t] = (_Float16)val;
                } else if (mode == MODE_RESID) {   // h += proj
                    size_t idx = (size_t)m * N + n;
                    out32[idx] = out32[idx] + val;
                } else {                           // exact GELU
                    float gl = 0.5f * val * (1.0f + erff(val * 0.70710678118654752f));
                    out16[(size_t)m * N + n] = (_Float16)gl;
                }
            }
        }
    }
}

// ---------------------------------------------------------------------------
// Flash attention: block = 8 waves, each wave owns 16 query rows.
// grid.x = B*H*(T/128). Online softmax over 32-key chunks, WMMA for both
// S = Q K^T (K-dim = d = 64) and Y += P V (K-dim = t = 32).
// q,k layout [B,T,H,D]; v layout [B,H*D,T] (transposed by GEMM epilogue).
// ---------------------------------------------------------------------------
__global__ __launch_bounds__(256) void attn_kernel(const _Float16* __restrict__ q,
                                                   const _Float16* __restrict__ k,
                                                   const _Float16* __restrict__ vt,
                                                   _Float16* __restrict__ y) {
    const int tid = threadIdx.x, w = tid >> 5, lane = tid & 31;
    const int g = lane >> 4, ln16 = lane & 15;
    int blk = blockIdx.x;
    int qb  = blk & 3;            // 4 q-blocks of 128 rows
    int bh  = blk >> 2;
    int b   = bh >> 4;            // H = 16
    int h   = bh & 15;
    int q0  = qb * 128 + w * 16;  // this wave's first query row

    __shared__ _Float16 ps[8][16 * 40];   // per-wave P staging (padded)
    _Float16* myps = ps[w];

    // Q A-fragments (held for the whole kv loop): rows = q0+ln16, K-dim = d
    const _Float16* qrow = q + ((size_t)(b * TT + q0 + ln16) * CC) + h * DH;
    H16 aq0, aq1;
    aq0.h[0] = *(const half8*)(qrow + g * 16);
    aq0.h[1] = *(const half8*)(qrow + g * 16 + 8);
    aq1.h[0] = *(const half8*)(qrow + 32 + g * 16);
    aq1.h[1] = *(const half8*)(qrow + 32 + g * 16 + 8);

    v8f yacc[4];
#pragma unroll
    for (int nt = 0; nt < 4; ++nt) yacc[nt] = (v8f){0,0,0,0,0,0,0,0};
    float m_run[8], l_run[8];
#pragma unroll
    for (int r = 0; r < 8; ++r) { m_run[r] = -3.4e38f; l_run[r] = 0.f; }

    const _Float16* kbase = k  + ((size_t)b * TT) * CC + h * DH;
    const _Float16* vbase = vt + ((size_t)b * CC + h * DH) * TT;

    for (int kv = 0; kv < TT; kv += 32) {
        // S tiles: n-tile 0 -> keys kv..kv+15, n-tile 1 -> kv+16..kv+31
        v8f s0 = (v8f){0,0,0,0,0,0,0,0};
        v8f s1 = (v8f){0,0,0,0,0,0,0,0};
        {
            const _Float16* kr0 = kbase + (size_t)(kv + ln16) * CC;
            const _Float16* kr1 = kbase + (size_t)(kv + 16 + ln16) * CC;
            half16 bk;
            bk = *(const half16*)(kr0 + g * 16);       s0 = wmma_f16(aq0.v, bk, s0);
            bk = *(const half16*)(kr0 + 32 + g * 16);  s0 = wmma_f16(aq1.v, bk, s0);
            bk = *(const half16*)(kr1 + g * 16);       s1 = wmma_f16(aq0.v, bk, s1);
            bk = *(const half16*)(kr1 + 32 + g * 16);  s1 = wmma_f16(aq1.v, bk, s1);
        }

        // online softmax per row (row M = g*8+r lives in one 16-lane half)
        float corr[8];
#pragma unroll
        for (int r = 0; r < 8; ++r) {
            float e0 = s0[r] * 0.125f;   // 1/sqrt(64)
            float e1 = s1[r] * 0.125f;
            float mx = fmaxf(e0, e1);
#pragma unroll
            for (int msk = 1; msk < 16; msk <<= 1) mx = fmaxf(mx, __shfl_xor(mx, msk, 32));
            float mnew = fmaxf(m_run[r], mx);
            float c  = expf(m_run[r] - mnew);
            float p0 = expf(e0 - mnew);
            float p1 = expf(e1 - mnew);
            float sm = p0 + p1;
#pragma unroll
            for (int msk = 1; msk < 16; msk <<= 1) sm += __shfl_xor(sm, msk, 32);
            l_run[r] = l_run[r] * c + sm;
            m_run[r] = mnew;
            corr[r]  = c;
            int row = g * 8 + r;
            myps[row * 40 + ln16]      = (_Float16)p0;
            myps[row * 40 + 16 + ln16] = (_Float16)p1;
        }
#pragma unroll
        for (int nt = 0; nt < 4; ++nt)
#pragma unroll
            for (int r = 0; r < 8; ++r) yacc[nt][r] *= corr[r];

        __syncthreads();   // re-layout P (accumulator -> A-fragment) via LDS

        H16 ap;
        ap.h[0] = *(const half8*)&myps[ln16 * 40 + g * 16];
        ap.h[1] = *(const half8*)&myps[ln16 * 40 + g * 16 + 8];
#pragma unroll
        for (int nt = 0; nt < 4; ++nt) {
            const _Float16* vr = vbase + (size_t)(nt * 16 + ln16) * TT + kv + g * 16;
            half16 bv = *(const half16*)vr;
            yacc[nt] = wmma_f16(ap.v, bv, yacc[nt]);
        }
        __syncthreads();
    }

    // normalize and store Y as f16 [B,T,H,D]
#pragma unroll
    for (int r = 0; r < 8; ++r) {
        float inv = 1.0f / l_run[r];
        int trow = q0 + g * 8 + r;
        _Float16* yr = y + ((size_t)(b * TT + trow) * CC) + h * DH;
#pragma unroll
        for (int nt = 0; nt < 4; ++nt)
            yr[nt * 16 + ln16] = (_Float16)(yacc[nt][r] * inv);
    }
}

// ---------------------------------------------------------------------------
// LayerNorm over C=1024, f16 output. One block (256 thr) per row.
// ---------------------------------------------------------------------------
__global__ __launch_bounds__(256) void ln_kernel(const float* __restrict__ x,
                                                 const float* __restrict__ gamma,
                                                 const float* __restrict__ beta,
                                                 _Float16* __restrict__ out) {
    int row = blockIdx.x, tid = threadIdx.x;
    const float* xr = x + (size_t)row * CC;
    float v[4], s = 0.f, s2 = 0.f;
#pragma unroll
    for (int i = 0; i < 4; ++i) {
        v[i] = xr[tid + 256 * i];
        s += v[i]; s2 += v[i] * v[i];
    }
#pragma unroll
    for (int off = 16; off >= 1; off >>= 1) {
        s  += __shfl_xor(s, off, 32);
        s2 += __shfl_xor(s2, off, 32);
    }
    __shared__ float r1[8], r2[8];
    if ((tid & 31) == 0) { r1[tid >> 5] = s; r2[tid >> 5] = s2; }
    __syncthreads();
    float S = 0.f, S2 = 0.f;
#pragma unroll
    for (int i = 0; i < 8; ++i) { S += r1[i]; S2 += r2[i]; }
    float mean = S * (1.0f / CC);
    float var  = S2 * (1.0f / CC) - mean * mean;
    float rs   = rsqrtf(var + 1e-5f);
#pragma unroll
    for (int i = 0; i < 4; ++i) {
        int c = tid + 256 * i;
        out[(size_t)row * CC + c] = (_Float16)((v[i] - mean) * rs * gamma[c] + beta[c]);
    }
}

// ---------------------------------------------------------------------------
// Final LN (last token only) + pooled head: out[b, 0..255]
// ---------------------------------------------------------------------------
__global__ __launch_bounds__(256) void head_kernel(const float* __restrict__ hbuf,
                                                   const float* __restrict__ gf,
                                                   const float* __restrict__ bf,
                                                   const float* __restrict__ Wp,
                                                   const float* __restrict__ bp,
                                                   float* __restrict__ out) {
    int b = blockIdx.x, tid = threadIdx.x;
    const float* xr = hbuf + ((size_t)(b * TT + (TT - 1))) * CC;
    __shared__ float xs[CC];
    __shared__ float r1[8], r2[8];
    float v[4], s = 0.f, s2 = 0.f;
#pragma unroll
    for (int i = 0; i < 4; ++i) {
        v[i] = xr[tid + 256 * i];
        s += v[i]; s2 += v[i] * v[i];
    }
#pragma unroll
    for (int off = 16; off >= 1; off >>= 1) {
        s  += __shfl_xor(s, off, 32);
        s2 += __shfl_xor(s2, off, 32);
    }
    if ((tid & 31) == 0) { r1[tid >> 5] = s; r2[tid >> 5] = s2; }
    __syncthreads();
    float S = 0.f, S2 = 0.f;
#pragma unroll
    for (int i = 0; i < 8; ++i) { S += r1[i]; S2 += r2[i]; }
    float mean = S * (1.0f / CC);
    float var  = S2 * (1.0f / CC) - mean * mean;
    float rs   = rsqrtf(var + 1e-5f);
#pragma unroll
    for (int i = 0; i < 4; ++i) {
        int c = tid + 256 * i;
        xs[c] = (v[i] - mean) * rs * gf[c] + bf[c];
    }
    __syncthreads();
    float acc = bp[tid];
    for (int kk = 0; kk < CC; ++kk) acc += xs[kk] * Wp[(size_t)kk * NOUT + tid];
    out[(size_t)b * NOUT + tid] = acc;
}

// ---------------------------------------------------------------------------
// Host orchestration
// ---------------------------------------------------------------------------
extern "C" void kernel_launch(void* const* d_in, const int* in_sizes, int n_in,
                              void* d_out, int out_size, void* d_ws, size_t ws_size,
                              hipStream_t stream) {
    (void)in_sizes; (void)n_in; (void)out_size;
    const float* x   = (const float*)d_in[0];
    const float* Wq  = (const float*)d_in[1];
    const float* bq  = (const float*)d_in[2];
    const float* Wk  = (const float*)d_in[3];
    const float* bk  = (const float*)d_in[4];
    const float* Wv  = (const float*)d_in[5];
    const float* bv  = (const float*)d_in[6];
    const float* Wo  = (const float*)d_in[7];
    const float* bo  = (const float*)d_in[8];
    const float* g1  = (const float*)d_in[9];
    const float* be1 = (const float*)d_in[10];
    const float* g2  = (const float*)d_in[11];
    const float* be2 = (const float*)d_in[12];
    const float* W1  = (const float*)d_in[13];
    const float* b1  = (const float*)d_in[14];
    const float* W2  = (const float*)d_in[15];
    const float* b2  = (const float*)d_in[16];
    const float* gf  = (const float*)d_in[17];
    const float* bf  = (const float*)d_in[18];
    const float* Wp  = (const float*)d_in[19];
    const float* bp  = (const float*)d_in[20];
    float* out = (float*)d_out;

    // workspace layout (all 256B aligned; total = 200 MB)
    char* ws = (char*)d_ws;
    size_t off = 0;
    auto take = [&](size_t bytes) { size_t o = off; off += (bytes + 255) & ~(size_t)255; return o; };
    float*    hbuf = (float*)   (ws + take((size_t)MM * CC * 4));
    _Float16* a16  = (_Float16*)(ws + take((size_t)MM * CC * 2));
    _Float16* q16  = (_Float16*)(ws + take((size_t)MM * CC * 2));
    _Float16* k16  = (_Float16*)(ws + take((size_t)MM * CC * 2));
    _Float16* vt16 = (_Float16*)(ws + take((size_t)MM * CC * 2));
    _Float16* y16  = (_Float16*)(ws + take((size_t)MM * CC * 2));
    _Float16* u16  = (_Float16*)(ws + take((size_t)MM * FF * 2));
    _Float16* pwq  = (_Float16*)(ws + take((size_t)CC * CC * 2));
    _Float16* pwk  = (_Float16*)(ws + take((size_t)CC * CC * 2));
    _Float16* pwv  = (_Float16*)(ws + take((size_t)CC * CC * 2));
    _Float16* pwo  = (_Float16*)(ws + take((size_t)CC * CC * 2));
    _Float16* pw1  = (_Float16*)(ws + take((size_t)CC * FF * 2));
    _Float16* pw2  = (_Float16*)(ws + take((size_t)FF * CC * 2));
    if (off > ws_size) return;   // workspace too small: nothing safe to do

    // h = x
    hipMemcpyAsync(hbuf, x, (size_t)MM * CC * sizeof(float), hipMemcpyDeviceToDevice, stream);

    const dim3 blk(256);
    const dim3 gemmCC(MM / 128, CC / 128);   // 64 x 8
    const dim3 gemmCF(MM / 128, FF / 128);   // 64 x 32

    for (int l = 0; l < LL; ++l) {
        const size_t wCC = (size_t)l * CC * CC;
        const size_t wCF = (size_t)l * CC * FF;

        // ln1 -> a
        ln_kernel<<<MM, blk, 0, stream>>>(hbuf, g1 + l * CC, be1 + l * CC, a16);

        // QKV projections
        pack_kernel<<<256, blk, 0, stream>>>(Wq + wCC, pwq, CC, CC);
        gemm_kernel<<<gemmCC, blk, 0, stream>>>(a16, pwq, bq + l * CC, CC, CC, MODE_F16, nullptr, q16);
        pack_kernel<<<256, blk, 0, stream>>>(Wk + wCC, pwk, CC, CC);
        gemm_kernel<<<gemmCC, blk, 0, stream>>>(a16, pwk, bk + l * CC, CC, CC, MODE_F16, nullptr, k16);
        pack_kernel<<<256, blk, 0, stream>>>(Wv + wCC, pwv, CC, CC);
        gemm_kernel<<<gemmCC, blk, 0, stream>>>(a16, pwv, bv + l * CC, CC, CC, MODE_VT, nullptr, vt16);

        // attention
        attn_kernel<<<BB * HH * (TT / 128), blk, 0, stream>>>(q16, k16, vt16, y16);

        // output projection + residual
        pack_kernel<<<256, blk, 0, stream>>>(Wo + wCC, pwo, CC, CC);
        gemm_kernel<<<gemmCC, blk, 0, stream>>>(y16, pwo, bo + l * CC, CC, CC, MODE_RESID, hbuf, nullptr);

        // ln2 -> a ; MLP
        ln_kernel<<<MM, blk, 0, stream>>>(hbuf, g2 + l * CC, be2 + l * CC, a16);
        pack_kernel<<<1024, blk, 0, stream>>>(W1 + wCF, pw1, CC, FF);
        gemm_kernel<<<gemmCF, blk, 0, stream>>>(a16, pw1, b1 + l * FF, CC, FF, MODE_GELU, nullptr, u16);
        pack_kernel<<<1024, blk, 0, stream>>>(W2 + wCF, pw2, FF, CC);
        gemm_kernel<<<gemmCC, blk, 0, stream>>>(u16, pw2, b2 + l * CC, FF, CC, MODE_RESID, hbuf, nullptr);
    }

    head_kernel<<<BB, blk, 0, stream>>>(hbuf, gf, bf, Wp, bp, out);
}
